// DyHeadBlock_11708080849520
// MI455X (gfx1250) — matrix operand order repaired and checked by reading, not compiled, not run on hardware
//
#include <hip/hip_runtime.h>
#include <math.h>

typedef __attribute__((ext_vector_type(16))) __bf16 v16bf;
typedef __attribute__((ext_vector_type(8)))  float  v8f;

#define CIN   256
#define KTAP  9
#define KTOT  (CIN*KTAP)   // 2304
#define KS    32           // K per WMMA step (bf16 16x16x32)
#define MT    32           // pixels per block tile (all HW sizes are multiples of 32)
#define BLOCK 256          // 8 waves (wave32)

union ABfrag { v16bf v; uint4 q[2]; };

__device__ __forceinline__ unsigned short f2bf(float f) {
  unsigned u = __builtin_bit_cast(unsigned, f);
  unsigned r = u + 0x7FFFu + ((u >> 16) & 1u);   // round-to-nearest-even
  return (unsigned short)(r >> 16);
}

// ---- weight conversion with K-permutation: out[o][tap*256+c] = in[o][c*9+tap]
__global__ void k_cvt_tr(const float* __restrict__ in,
                         unsigned short* __restrict__ out,
                         int rows_in, int rows_out) {
  int i = blockIdx.x * blockDim.x + threadIdx.x;
  int n = rows_out * KTOT;
  if (i >= n) return;
  int o = i / KTOT;
  int r = i - o * KTOT;    // r = tap*256 + c
  int tap = r >> 8;
  int c   = r & 255;
  out[i] = (o < rows_in) ? f2bf(in[(size_t)o * KTOT + c * KTAP + tap])
                         : (unsigned short)0;
}

// ---------------- bilinear resize (align_corners=True) ----------------
__global__ void k_resize(const float* __restrict__ in, int Hi, int Wi,
                         float* __restrict__ out, int Ho, int Wo, int NC) {
  int i = blockIdx.x * blockDim.x + threadIdx.x;
  int total = NC * Ho * Wo;
  if (i >= total) return;
  int xo = i % Wo;
  int yo = (i / Wo) % Ho;
  int n  = i / (Wo * Ho);
  float sy = (Ho > 1) ? (float)(Hi - 1) / (float)(Ho - 1) : 0.f;
  float sx = (Wo > 1) ? (float)(Wi - 1) / (float)(Wo - 1) : 0.f;
  float fy = yo * sy, fx = xo * sx;
  int yl = (int)floorf(fy), xl = (int)floorf(fx);
  int yh = min(yl + 1, Hi - 1), xh = min(xl + 1, Wi - 1);
  float wy = fy - (float)yl, wx = fx - (float)xl;
  const float* p = in + (size_t)n * Hi * Wi;
  float v = p[yl * Wi + xl] * (1.f - wy) * (1.f - wx)
          + p[yl * Wi + xh] * (1.f - wy) * wx
          + p[yh * Wi + xl] * wy * (1.f - wx)
          + p[yh * Wi + xh] * wy * wx;
  out[i] = v;
}

// ---------------- offset conv: 3x3 pad1, 256->81 (WMMA, N padded to 128) ----
__global__ __launch_bounds__(BLOCK) void k_conv_off(
    const float* __restrict__ x, int H, int W,
    const unsigned short* __restrict__ wq,   // [128][KTOT] bf16, tap-major K
    const float* __restrict__ bias,          // [81]
    float* __restrict__ om)                  // [B][81][H*W]
{
  __shared__ unsigned short Alds[2][MT * KS];
  __shared__ float Cwt[MT][KTAP];
  __shared__ int   Cix[MT][KTAP];

  const int b    = blockIdx.y;
  const int tile = blockIdx.x;
  const int HW   = H * W;
  const int tid  = threadIdx.x;
  const int lane = tid & 31;
  const int wv   = tid >> 5;
  const int msub = wv & 1;
  const int nbase = (wv >> 1) * 32;      // 4 wave-groups cover N=128
  const float* xb = x + (size_t)b * CIN * HW;

  // tap table: branchless zero-pad (clamped index, 0/1 weight)
  for (int j = tid; j < MT * KTAP; j += BLOCK) {
    int pix = j / KTAP, k = j - KTAP * pix;
    int p = tile * MT + pix;
    int y = p / W, xq = p % W;
    int ky = k / 3, kx = k - 3 * ky;
    int yy = y - 1 + ky, xx = xq - 1 + kx;
    bool valid = (unsigned)yy < (unsigned)H && (unsigned)xx < (unsigned)W;
    int yc = min(max(yy, 0), H - 1), xc = min(max(xx, 0), W - 1);
    Cwt[pix][k] = valid ? 1.f : 0.f;
    Cix[pix][k] = yc * W + xc;
  }
  __syncthreads();

  const int spix = tid >> 3;             // 0..31
  const int scc  = (tid & 7) * 4;        // channel sub-offset

  v8f acc0 = {}; v8f acc1 = {};

  for (int k0 = 0; k0 < KTOT; k0 += KS) {
    int buf = (k0 >> 5) & 1;
    int ktap = k0 >> 8;                  // constant tap this step
    int cb   = (k0 & 255) + scc;
    float wgt = Cwt[spix][ktap];
    int   idx = Cix[spix][ktap];
    const float* xc = xb + (size_t)cb * HW + idx;
    unsigned short h0 = f2bf(wgt * xc[0]);
    unsigned short h1 = f2bf(wgt * xc[(size_t)HW]);
    unsigned short h2 = f2bf(wgt * xc[(size_t)2 * HW]);
    unsigned short h3 = f2bf(wgt * xc[(size_t)3 * HW]);
    unsigned pack0 = (unsigned)h0 | ((unsigned)h1 << 16);
    unsigned pack1 = (unsigned)h2 | ((unsigned)h3 << 16);
    *(uint2*)&Alds[buf][spix * KS + scc] = make_uint2(pack0, pack1);
    __syncthreads();

    ABfrag a;
    {
      int row  = msub * 16 + (lane & 15);
      int half = lane >> 4;
      const unsigned short* base = &Alds[buf][row * KS + half * 8];
      a.q[0] = *(const uint4*)(base);
      a.q[1] = *(const uint4*)(base + 16);
    }
#pragma unroll
    for (int sub = 0; sub < 2; sub++) {
      ABfrag bm;
      int o = nbase + sub * 16 + (lane & 15);
      const unsigned short* wp = wq + (size_t)o * KTOT + k0 + ((lane >> 4) << 3);
      bm.q[0] = *(const uint4*)(wp);
      bm.q[1] = *(const uint4*)(wp + 16);
      if (sub == 0)
        acc0 = __builtin_amdgcn_wmma_f32_16x16x32_bf16(false, a.v, false, bm.v, (short)0, acc0, false, false);
      else
        acc1 = __builtin_amdgcn_wmma_f32_16x16x32_bf16(false, a.v, false, bm.v, (short)0, acc1, false, false);
    }
  }

  // epilogue: p always in range (HW % 32 == 0); only o<81 is real
  const int p0 = tile * MT + msub * 16 + (lane >> 4) * 8;
#pragma unroll
  for (int sub = 0; sub < 2; sub++) {
    v8f acc = sub ? acc1 : acc0;
    int o = nbase + sub * 16 + (lane & 15);
    if (o < 81) {
      float bo = bias[o];
      float* op = om + ((size_t)(b * 81 + o)) * HW + p0;
#pragma unroll
      for (int r = 0; r < 8; r++) op[r] = acc[r] + bo;
    }
  }
}

// ---------------- modulated deformable conv (WMMA implicit GEMM) ----------
__global__ __launch_bounds__(BLOCK) void k_mdcn(
    const float* __restrict__ x, int Hs, int Ws,
    const float* __restrict__ om, int Ho, int Wo,
    int off_base, int mask_base, int stride,
    const unsigned short* __restrict__ wq,   // [256][KTOT] bf16, tap-major K
    float* __restrict__ out)                 // [B][256][Ho*Wo]
{
  __shared__ unsigned short Alds[2][MT * KS];
  __shared__ float4 Twt[MT][KTAP];
  __shared__ int4   Tix[MT][KTAP];

  const int b    = blockIdx.y;
  const int tile = blockIdx.x;
  const int HWo  = Ho * Wo;
  const int HWs  = Hs * Ws;
  const int tid  = threadIdx.x;
  const int lane = tid & 31;
  const int wv   = tid >> 5;
  const int msub = wv & 1;
  const int nbase = (wv >> 1) * 64;     // wave covers N range of 64 (4 subtiles)
  const float* xb = x + (size_t)b * CIN * HWs;

  // ---- tap table: 32 pixels x 9 taps, mask folded into 4 corner weights ----
  for (int j = tid; j < MT * KTAP; j += BLOCK) {
    int pix = j / KTAP, k = j - KTAP * pix;
    int p = tile * MT + pix;
    int y = p / Wo, xq = p % Wo;
    float dy = om[((size_t)(b * 81 + off_base + 2 * k)) * HWo + p];
    float dx = om[((size_t)(b * 81 + off_base + 2 * k + 1)) * HWo + p];
    float mv = om[((size_t)(b * 81 + mask_base + k)) * HWo + p];
    mv = 1.f / (1.f + expf(-mv));
    int ky = k / 3, kx = k - 3 * ky;
    float py = (float)(y * stride - 1 + ky) + dy;
    float px = (float)(xq * stride - 1 + kx) + dx;
    float y0f = floorf(py), x0f = floorf(px);
    float wy = py - y0f, wx = px - x0f;
    int y0 = (int)y0f, x0 = (int)x0f;
    int y1 = y0 + 1, x1 = x0 + 1;
    bool vy0 = (unsigned)y0 < (unsigned)Hs, vy1 = (unsigned)y1 < (unsigned)Hs;
    bool vx0 = (unsigned)x0 < (unsigned)Ws, vx1 = (unsigned)x1 < (unsigned)Ws;
    int y0c = min(max(y0, 0), Hs - 1), y1c = min(max(y1, 0), Hs - 1);
    int x0c = min(max(x0, 0), Ws - 1), x1c = min(max(x1, 0), Ws - 1);
    float4 w4;
    w4.x = (vy0 && vx0) ? (1.f - wy) * (1.f - wx) * mv : 0.f;
    w4.y = (vy0 && vx1) ? (1.f - wy) * wx * mv : 0.f;
    w4.z = (vy1 && vx0) ? wy * (1.f - wx) * mv : 0.f;
    w4.w = (vy1 && vx1) ? wy * wx * mv : 0.f;
    int4 i4;
    i4.x = y0c * Ws + x0c;
    i4.y = y0c * Ws + x1c;
    i4.z = y1c * Ws + x0c;
    i4.w = y1c * Ws + x1c;
    Twt[pix][k] = w4;
    Tix[pix][k] = i4;
  }
  __syncthreads();

  const int spix = tid >> 3;
  const int scc  = (tid & 7) * 4;

  v8f acc[4] = {v8f{}, v8f{}, v8f{}, v8f{}};

  for (int k0 = 0; k0 < KTOT; k0 += KS) {
    int buf = (k0 >> 5) & 1;
    int ktap = k0 >> 8;                  // constant tap this step
    int cb   = (k0 & 255) + scc;
    float4 w4 = Twt[spix][ktap];
    int4   i4 = Tix[spix][ktap];
    unsigned short h[4];
#pragma unroll
    for (int i = 0; i < 4; i++) {
      const float* xc = xb + (size_t)(cb + i) * HWs;
      float v = w4.x * xc[i4.x] + w4.y * xc[i4.y] +
                w4.z * xc[i4.z] + w4.w * xc[i4.w];
      h[i] = f2bf(v);
    }
    unsigned pack0 = (unsigned)h[0] | ((unsigned)h[1] << 16);
    unsigned pack1 = (unsigned)h[2] | ((unsigned)h[3] << 16);
    *(uint2*)&Alds[buf][spix * KS + scc] = make_uint2(pack0, pack1);
    __syncthreads();

    ABfrag a;
    {
      int row  = msub * 16 + (lane & 15);
      int half = lane >> 4;
      const unsigned short* base = &Alds[buf][row * KS + half * 8];
      a.q[0] = *(const uint4*)(base);
      a.q[1] = *(const uint4*)(base + 16);
    }
#pragma unroll
    for (int sub = 0; sub < 4; sub++) {
      ABfrag bm;
      int o = nbase + sub * 16 + (lane & 15);
      const unsigned short* wp = wq + (size_t)o * KTOT + k0 + ((lane >> 4) << 3);
      bm.q[0] = *(const uint4*)(wp);
      bm.q[1] = *(const uint4*)(wp + 16);
      acc[sub] = __builtin_amdgcn_wmma_f32_16x16x32_bf16(false, a.v, false, bm.v, (short)0, acc[sub], false, false);
    }
  }

  // epilogue: p always in range (HWo % 32 == 0), o always < 256
  const int p0 = tile * MT + msub * 16 + (lane >> 4) * 8;
#pragma unroll
  for (int sub = 0; sub < 4; sub++) {
    int o = nbase + sub * 16 + (lane & 15);
    float* op = out + ((size_t)(b * CIN + o)) * HWo + p0;
#pragma unroll
    for (int r = 0; r < 8; r++) op[r] = acc[sub][r];
  }
}

// ---------------- group norm (16 groups of 16 contiguous channels) ---------
__global__ void k_gn_stats(const float* __restrict__ v, int HW,
                           float* __restrict__ stats) {  // grid = B*16
  int bg = blockIdx.x;
  const float* p = v + (size_t)bg * 16 * HW;
  int n = 16 * HW;
  float s = 0.f, s2 = 0.f;
  for (int i = threadIdx.x; i < n; i += blockDim.x) {
    float t = p[i]; s += t; s2 += t * t;
  }
  __shared__ float rs[256], rs2[256];
  rs[threadIdx.x] = s; rs2[threadIdx.x] = s2;
  __syncthreads();
  for (int st = 128; st > 0; st >>= 1) {
    if (threadIdx.x < st) {
      rs[threadIdx.x]  += rs[threadIdx.x + st];
      rs2[threadIdx.x] += rs2[threadIdx.x + st];
    }
    __syncthreads();
  }
  if (threadIdx.x == 0) {
    float inv = 1.f / (float)n;
    float mean = rs[0] * inv;
    float var  = rs2[0] * inv - mean * mean;
    stats[bg * 2]     = mean;
    stats[bg * 2 + 1] = rsqrtf(var + 1e-5f);
  }
}

__global__ void k_gn_accum(const float* __restrict__ v,
                           const float* __restrict__ stats,
                           const float* __restrict__ gamma,
                           const float* __restrict__ beta,
                           float* __restrict__ dst, int HW, int first, int total) {
  int i = blockIdx.x * blockDim.x + threadIdx.x;
  if (i >= total) return;
  int c  = (i / HW) & 255;
  int bg = i / (16 * HW);
  float mean = stats[bg * 2], rstd = stats[bg * 2 + 1];
  float y = (v[i] - mean) * rstd * gamma[c] + beta[c];
  dst[i] = first ? y : dst[i] + y;
}

__global__ void k_scale(float* __restrict__ p, int n, float s) {
  int i = blockIdx.x * blockDim.x + threadIdx.x;
  if (i < n) p[i] *= s;
}

// ---------------- host orchestration ----------------
static inline size_t al256(size_t x) { return (x + 255) & ~(size_t)255; }

extern "C" void kernel_launch(void* const* d_in, const int* in_sizes, int n_in,
                              void* d_out, int out_size, void* d_ws, size_t ws_size,
                              hipStream_t stream) {
  (void)in_sizes; (void)n_in; (void)out_size; (void)ws_size;
  const float* x0     = (const float*)d_in[0];
  const float* x1     = (const float*)d_in[1];
  const float* x2     = (const float*)d_in[2];
  const float* w_off  = (const float*)d_in[3];
  const float* b_off  = (const float*)d_in[4];
  const float* w_mid  = (const float*)d_in[5];
  const float* g_mid  = (const float*)d_in[6];
  const float* be_mid = (const float*)d_in[7];
  const float* w_low  = (const float*)d_in[8];
  const float* g_low  = (const float*)d_in[9];
  const float* be_low = (const float*)d_in[10];
  const float* w_high = (const float*)d_in[11];
  const float* g_high = (const float*)d_in[12];
  const float* be_high= (const float*)d_in[13];

  const int B = 2;
  const int H0 = 96, W0 = 96, H1 = 48, W1 = 48, H2 = 24, W2 = 24;
  const int hw0 = H0 * W0, hw1 = H1 * W1, hw2 = H2 * W2;
  const size_t N0 = (size_t)B * CIN * hw0;   // 4718592
  const size_t N1 = (size_t)B * CIN * hw1;   // 1179648
  const size_t N2 = (size_t)B * CIN * hw2;   //  294912

  char* w = (char*)d_ws;
  size_t off = 0;
  auto take = [&](size_t bytes) { void* p = w + off; off = al256(off + bytes); return p; };
  unsigned short* wq_off  = (unsigned short*)take((size_t)128 * KTOT * 2);
  unsigned short* wq_mid  = (unsigned short*)take((size_t)256 * KTOT * 2);
  unsigned short* wq_low  = (unsigned short*)take((size_t)256 * KTOT * 2);
  unsigned short* wq_high = (unsigned short*)take((size_t)256 * KTOT * 2);
  float* om0  = (float*)take((size_t)B * 81 * hw0 * 4);
  float* om1  = (float*)take((size_t)B * 81 * hw1 * 4);
  float* om2  = (float*)take((size_t)B * 81 * hw2 * 4);
  float* rs0  = (float*)take(N0 * 4);   // x1 upsampled to 96x96
  float* rs1  = (float*)take(N1 * 4);   // x2 upsampled to 48x48
  float* tmp  = (float*)take(N0 * 4);   // branch scratch (max size)
  float* stats = (float*)take(64 * 4);

  float* out0 = (float*)d_out;
  float* out1 = out0 + N0;
  float* out2 = out1 + N1;

  dim3 blk(BLOCK);

  // weights -> bf16 with tap-major K permutation
  {
    int n = 256 * KTOT;
    k_cvt_tr<<<dim3((n + BLOCK - 1) / BLOCK), blk, 0, stream>>>(w_mid,  wq_mid,  256, 256);
    k_cvt_tr<<<dim3((n + BLOCK - 1) / BLOCK), blk, 0, stream>>>(w_low,  wq_low,  256, 256);
    k_cvt_tr<<<dim3((n + BLOCK - 1) / BLOCK), blk, 0, stream>>>(w_high, wq_high, 256, 256);
    int np = 128 * KTOT;
    k_cvt_tr<<<dim3((np + BLOCK - 1) / BLOCK), blk, 0, stream>>>(w_off, wq_off, 81, 128);
  }

  // upsample next-level features (align_corners=True)
  k_resize<<<dim3((int)((N0 + BLOCK - 1) / BLOCK)), blk, 0, stream>>>(x1, H1, W1, rs0, H0, W0, B * CIN);
  k_resize<<<dim3((int)((N1 + BLOCK - 1) / BLOCK)), blk, 0, stream>>>(x2, H2, W2, rs1, H1, W1, B * CIN);

  // offset convs
  k_conv_off<<<dim3(hw0 / MT, B), blk, 0, stream>>>(x0, H0, W0, wq_off, b_off, om0);
  k_conv_off<<<dim3(hw1 / MT, B), blk, 0, stream>>>(x1, H1, W1, wq_off, b_off, om1);
  k_conv_off<<<dim3(hw2 / MT, B), blk, 0, stream>>>(x2, H2, W2, wq_off, b_off, om2);

  auto branch = [&](const float* src, int Hs, int Ws, const float* om, int Ho, int Wo,
                    int obase, int mbase, int stride, const unsigned short* wq,
                    const float* gma, const float* bta, float* dst, int first) {
    int hwo = Ho * Wo;
    int total = B * CIN * hwo;
    k_mdcn<<<dim3(hwo / MT, B), blk, 0, stream>>>(
        src, Hs, Ws, om, Ho, Wo, obase, mbase, stride, wq, tmp);
    k_gn_stats<<<dim3(B * 16), blk, 0, stream>>>(tmp, hwo, stats);
    k_gn_accum<<<dim3((total + BLOCK - 1) / BLOCK), blk, 0, stream>>>(
        tmp, stats, gma, bta, dst, hwo, first, total);
  };

  // ---- level 0: mid + high, /2 ----
  branch(x0,  H0, W0, om0, H0, W0,  0, 54, 1, wq_mid,  g_mid,  be_mid,  out0, 1);
  branch(rs0, H0, W0, om0, H0, W0, 36, 72, 1, wq_high, g_high, be_high, out0, 0);
  k_scale<<<dim3((int)((N0 + BLOCK - 1) / BLOCK)), blk, 0, stream>>>(out0, (int)N0, 0.5f);

  // ---- level 1: mid + low(stride2 from x0) + high, /3 ----
  branch(x1,  H1, W1, om1, H1, W1,  0, 54, 1, wq_mid,  g_mid,  be_mid,  out1, 1);
  branch(x0,  H0, W0, om1, H1, W1, 18, 63, 2, wq_low,  g_low,  be_low,  out1, 0);
  branch(rs1, H1, W1, om1, H1, W1, 36, 72, 1, wq_high, g_high, be_high, out1, 0);
  k_scale<<<dim3((int)((N1 + BLOCK - 1) / BLOCK)), blk, 0, stream>>>(out1, (int)N1, 1.f / 3.f);

  // ---- level 2: mid + low(stride2 from x1), /2 ----
  branch(x2,  H2, W2, om2, H2, W2,  0, 54, 1, wq_mid,  g_mid,  be_mid,  out2, 1);
  branch(x1,  H1, W1, om2, H2, W2, 18, 63, 2, wq_low,  g_low,  be_low,  out2, 0);
  k_scale<<<dim3((int)((N2 + BLOCK - 1) / BLOCK)), blk, 0, stream>>>(out2, (int)N2, 0.5f);
}
